// EdgeGuidedFusionLoss_25529285607877
// MI455X (gfx1250) — compile-verified, matrix operand order checked by reading, not run on hardware
//
#include <hip/hip_runtime.h>

#define HH 512
#define WW 512
#define BB 16
#define HWSZ (HH*WW)          // 262144
#define NTOT (BB*HWSZ)        // 4194304
#define EPSF 1e-8f

// ---- ws layout (float offsets) ----
#define ACC_OFF   0      // [0]=l1m [1]=l1p [2]=petsal [3]=ssim_m [4]=ssim_p [5]=gfull [6]=ghalf [7]=aux [8]=sot [9]=sink
#define PMIN_OFF  16
#define PMAX_OFF  32
#define SMY_OFF   48
#define SMM_OFF   64
#define SMYH_OFF  80
#define SMMH_OFF  96
#define SY_OFF    112
#define SQ_OFF    128
#define HISTP_OFF 256
#define NPROJ 64
#define NBIN  512
#define HSZ   (BB*NPROJ*NBIN)            // 524288
#define HISTQ_OFF (HISTP_OFF + HSZ)      // 524544
#define HIST_END  (HISTQ_OFF + HSZ)      // 1048832
#define KS 512
#define XS_OFF  HIST_END
#define XT_OFF  (XS_OFF + BB*KS*2)
#define WA_OFF  (XT_OFF + BB*KS*2)
#define WB_OFF  (WA_OFF + BB*KS)
#define C_OFF   (WB_OFF + BB*KS)
// C size = BB*KS*KS = 4194304 floats; total ws ~21.2 MB

typedef __attribute__((ext_vector_type(16))) _Float16 v16h;
typedef __attribute__((ext_vector_type(8)))  float    v8f;

// ---- CDNA5 async global->LDS path (builtin exists on this toolchain; b32 takes int*) ----
#if defined(__has_builtin)
#if __has_builtin(__builtin_amdgcn_global_load_async_to_lds_b32)
#define HAVE_ASYNC_LDS 1
#endif
#endif

__device__ __forceinline__ void async_copy_b32(const float* gsrc, float* ldst) {
#if defined(HAVE_ASYNC_LDS)
    __builtin_amdgcn_global_load_async_to_lds_b32(
        (__attribute__((address_space(1))) int*)(int*)(void*)gsrc,
        (__attribute__((address_space(3))) int*)(int*)(void*)ldst, 0, 0);
#else
    *ldst = *gsrc;
#endif
}
__device__ __forceinline__ void wait_async0() {
#if defined(HAVE_ASYNC_LDS)
#if __has_builtin(__builtin_amdgcn_s_wait_asynccnt)
    __builtin_amdgcn_s_wait_asynccnt(0);
#else
    asm volatile("s_wait_asynccnt 0" ::: "memory");
#endif
#endif
}

// ---------------- helpers ----------------
__device__ __forceinline__ float ldpad(const float* img, int b, int r, int c) {
    if (r < 0 || r >= HH || c < 0 || c >= WW) return 0.f;
    return img[(size_t)b*HWSZ + r*WW + c];
}
__device__ __forceinline__ float poolld(const float* img, int b, int r, int c) {
    if (r < 0 || r >= HH/2 || c < 0 || c >= WW/2) return 0.f;
    const float* p = img + (size_t)b*HWSZ + (2*r)*WW + 2*c;
    return 0.25f*(p[0] + p[1] + p[WW] + p[WW+1]);
}
__device__ __forceinline__ float sobel_mag(const float* img, int b, int r, int c) {
    float a00=ldpad(img,b,r-1,c-1), a01=ldpad(img,b,r-1,c), a02=ldpad(img,b,r-1,c+1);
    float a10=ldpad(img,b,r,  c-1),                          a12=ldpad(img,b,r,  c+1);
    float a20=ldpad(img,b,r+1,c-1), a21=ldpad(img,b,r+1,c), a22=ldpad(img,b,r+1,c+1);
    float gx = a00 - a02 + 2.f*(a10 - a12) + a20 - a22;
    float gy = a00 + 2.f*a01 + a02 - a20 - 2.f*a21 - a22;
    return sqrtf(gx*gx + gy*gy + 1e-6f);
}
__device__ __forceinline__ float sobel_mag_h(const float* img, int b, int r, int c) {
    float a00=poolld(img,b,r-1,c-1), a01=poolld(img,b,r-1,c), a02=poolld(img,b,r-1,c+1);
    float a10=poolld(img,b,r,  c-1),                           a12=poolld(img,b,r,  c+1);
    float a20=poolld(img,b,r+1,c-1), a21=poolld(img,b,r+1,c), a22=poolld(img,b,r+1,c+1);
    float gx = a00 - a02 + 2.f*(a10 - a12) + a20 - a22;
    float gy = a00 + 2.f*a01 + a02 - a20 - 2.f*a21 - a22;
    return sqrtf(gx*gx + gy*gy + 1e-6f);
}
__device__ __forceinline__ float blockReduceSum(float v, float* sb) {
    int t = threadIdx.x; sb[t] = v; __syncthreads();
    for (int s = blockDim.x >> 1; s > 0; s >>= 1) { if (t < s) sb[t] += sb[t+s]; __syncthreads(); }
    float r = sb[0]; __syncthreads(); return r;
}
__device__ __forceinline__ float blockReduceMax(float v, float* sb) {
    int t = threadIdx.x; sb[t] = v; __syncthreads();
    for (int s = blockDim.x >> 1; s > 0; s >>= 1) { if (t < s) sb[t] = fmaxf(sb[t], sb[t+s]); __syncthreads(); }
    float r = sb[0]; __syncthreads(); return r;
}
__device__ __forceinline__ float blockReduceMin(float v, float* sb) {
    int t = threadIdx.x; sb[t] = v; __syncthreads();
    for (int s = blockDim.x >> 1; s > 0; s >>= 1) { if (t < s) sb[t] = fminf(sb[t], sb[t+s]); __syncthreads(); }
    float r = sb[0]; __syncthreads(); return r;
}
__device__ __forceinline__ unsigned hash_u(unsigned h) {
    h ^= h >> 16; h *= 0x7feb352du; h ^= h >> 15; h *= 0x846ca68bu; h ^= h >> 16; return h;
}

// ---------------- init ----------------
__global__ void init_kernel(float* ws) {
    unsigned* wu = (unsigned*)ws;
    for (size_t i = (size_t)blockIdx.x*blockDim.x + threadIdx.x; i < (size_t)HIST_END;
         i += (size_t)gridDim.x*blockDim.x) {
        if (i >= PMIN_OFF && i < PMIN_OFF+16) wu[i] = 0x7F800000u;  // +inf for pet-min
        else ws[i] = 0.f;
    }
}

// ---------------- pass A full: pet min/max, prob sums, sobel maxima ----------------
__global__ __launch_bounds__(256) void passA_full_kernel(const float* y, const float* mri,
                                                         const float* pet, const float* mask,
                                                         float* ws) {
    int b = blockIdx.y;
    unsigned* wu = (unsigned*)ws;
    float lmin = 3.0e38f, lmax = 0.f, lsy = 0.f, lsq = 0.f, lmy = 0.f, lmm = 0.f;
    for (int k = 0; k < 16; k++) {
        int i = blockIdx.x*4096 + k*256 + threadIdx.x;
        int r = i >> 9, c = i & 511;
        size_t idx = (size_t)b*HWSZ + i;
        __builtin_prefetch(&y[idx + 256], 0, 1);
        float pv = pet[idx]; lmin = fminf(lmin, pv); lmax = fmaxf(lmax, pv);
        float yv = y[idx];   lsy += fmaxf(yv, 0.f) + EPSF;
        float qv = mask[idx]; lsq += fminf(fmaxf(qv, 0.f), 1.f) + EPSF;
        lmy = fmaxf(lmy, sobel_mag(y,   b, r, c));
        lmm = fmaxf(lmm, sobel_mag(mri, b, r, c));
    }
    __shared__ float sb[256];
    float v;
    v = blockReduceSum(lsy, sb); if (!threadIdx.x) atomicAdd(&ws[SY_OFF+b], v);
    v = blockReduceSum(lsq, sb); if (!threadIdx.x) atomicAdd(&ws[SQ_OFF+b], v);
    v = blockReduceMax(lmy, sb); if (!threadIdx.x) atomicMax(&wu[SMY_OFF+b], __float_as_uint(v));
    v = blockReduceMax(lmm, sb); if (!threadIdx.x) atomicMax(&wu[SMM_OFF+b], __float_as_uint(v));
    v = blockReduceMax(lmax, sb); if (!threadIdx.x) atomicMax(&wu[PMAX_OFF+b], __float_as_uint(v));
    v = blockReduceMin(lmin, sb); if (!threadIdx.x) atomicMin(&wu[PMIN_OFF+b], __float_as_uint(v));
}

// ---------------- pass A half: pooled sobel maxima ----------------
__global__ __launch_bounds__(256) void passA_half_kernel(const float* y, const float* mri, float* ws) {
    int b = blockIdx.y;
    unsigned* wu = (unsigned*)ws;
    float lmy = 0.f, lmm = 0.f;
    for (int k = 0; k < 16; k++) {
        int i = blockIdx.x*4096 + k*256 + threadIdx.x;   // 65536 px / image
        int r = i >> 8, c = i & 255;
        lmy = fmaxf(lmy, sobel_mag_h(y,   b, r, c));
        lmm = fmaxf(lmm, sobel_mag_h(mri, b, r, c));
    }
    __shared__ float sb[256];
    float v;
    v = blockReduceMax(lmy, sb); if (!threadIdx.x) atomicMax(&wu[SMYH_OFF+b], __float_as_uint(v));
    v = blockReduceMax(lmm, sb); if (!threadIdx.x) atomicMax(&wu[SMMH_OFF+b], __float_as_uint(v));
}

// ---------------- pass B full: l1, pet-saliency, grad-consistency, aux ----------------
__global__ __launch_bounds__(256) void passB_full_kernel(const float* y, const float* mri,
                                                         const float* pet, const float* ehat,
                                                         float* ws) {
    int b = blockIdx.y;
    unsigned* wu = (unsigned*)ws;
    float myf = fmaxf(__uint_as_float(wu[SMY_OFF+b]), 1e-6f);
    float mmf = fmaxf(__uint_as_float(wu[SMM_OFF+b]), 1e-6f);
    float pmn = __uint_as_float(wu[PMIN_OFF+b]);
    float pmx = __uint_as_float(wu[PMAX_OFF+b]);
    float pinv = 1.f/(pmx - pmn + 1e-6f);
    float a_l1m=0.f, a_l1p=0.f, a_sal=0.f, a_g=0.f, a_aux=0.f;
    for (int k = 0; k < 16; k++) {
        int i = blockIdx.x*4096 + k*256 + threadIdx.x;
        int r = i >> 9, c = i & 511;
        size_t idx = (size_t)b*HWSZ + i;
        __builtin_prefetch(&mri[idx + 256], 0, 1);
        float yv = y[idx], mv = mri[idx], pv = pet[idx], ev = ehat[idx];
        a_l1m += fabsf(yv - mv);
        float dp = fabsf(yv - pv);
        a_l1p += dp;
        float pn = (pv - pmn)*pinv;
        float sg = 1.f/(1.f + __expf(-6.f*(pn - 0.4f)));
        a_sal += sg*dp;
        float syn = fminf(sobel_mag(y,   b, r, c)/myf, 1.f);
        float smn = fminf(sobel_mag(mri, b, r, c)/mmf, 1.f);
        float d = syn - smn;
        a_g   += sqrtf(d*d + 1e-6f);
        a_aux += fabsf(ev - smn);
    }
    __shared__ float sb[256];
    float v;
    v = blockReduceSum(a_l1m, sb); if (!threadIdx.x) atomicAdd(&ws[0], v);
    v = blockReduceSum(a_l1p, sb); if (!threadIdx.x) atomicAdd(&ws[1], v);
    v = blockReduceSum(a_sal, sb); if (!threadIdx.x) atomicAdd(&ws[2], v);
    v = blockReduceSum(a_g,   sb); if (!threadIdx.x) atomicAdd(&ws[5], v);
    v = blockReduceSum(a_aux, sb); if (!threadIdx.x) atomicAdd(&ws[7], v);
}

// ---------------- pass B half: grad-consistency at half res ----------------
__global__ __launch_bounds__(256) void passB_half_kernel(const float* y, const float* mri, float* ws) {
    int b = blockIdx.y;
    unsigned* wu = (unsigned*)ws;
    float myf = fmaxf(__uint_as_float(wu[SMYH_OFF+b]), 1e-6f);
    float mmf = fmaxf(__uint_as_float(wu[SMMH_OFF+b]), 1e-6f);
    float a_g = 0.f;
    for (int k = 0; k < 16; k++) {
        int i = blockIdx.x*4096 + k*256 + threadIdx.x;
        int r = i >> 8, c = i & 255;
        float syn = fminf(sobel_mag_h(y,   b, r, c)/myf, 1.f);
        float smn = fminf(sobel_mag_h(mri, b, r, c)/mmf, 1.f);
        float d = syn - smn;
        a_g += sqrtf(d*d + 1e-6f);
    }
    __shared__ float sb[256];
    float v = blockReduceSum(a_g, sb);
    if (!threadIdx.x) atomicAdd(&ws[6], v);
}

// ---------------- SSIM: separable 11x11 Gaussian, tiled in LDS ----------------
__global__ __launch_bounds__(256) void ssim_kernel(const float* y, const float* mri, const float* pet,
                                                   float* ws) {
    const int OUTSZ = 502;
    int z = blockIdx.z;
    int b = z >> 1, pair = z & 1;
    const float* aimg = y;
    const float* bimg = pair ? pet : mri;
    int ox = blockIdx.x*16, oy = blockIdx.y*16;

    float w[11];
    {
        float s = 0.f;
        #pragma unroll
        for (int i = 0; i < 11; i++) { float cc = (float)i - 5.f; w[i] = __expf(-(cc*cc)/4.5f); s += w[i]; }
        float inv = 1.f/s;
        #pragma unroll
        for (int i = 0; i < 11; i++) w[i] *= inv;
    }

    __shared__ float pa[26][27];
    __shared__ float pb[26][27];
    __shared__ float hb[5][26][16];

    // async global->LDS patch staging (ASYNCcnt-tracked when available)
    for (int p = threadIdx.x; p < 26*26; p += 256) {
        int pr = p/26, pc = p%26;
        int gr = min(oy + pr, HH-1), gc = min(ox + pc, WW-1);
        size_t idx = (size_t)b*HWSZ + gr*WW + gc;
        async_copy_b32(&aimg[idx], &pa[pr][pc]);
        async_copy_b32(&bimg[idx], &pb[pr][pc]);
    }
    wait_async0();
    __syncthreads();

    // horizontal pass: 26 rows x 16 cols, 5 products
    for (int p = threadIdx.x; p < 26*16; p += 256) {
        int r = p >> 4, c = p & 15;
        float s0=0.f, s1=0.f, s2=0.f, s3=0.f, s4=0.f;
        #pragma unroll
        for (int k = 0; k < 11; k++) {
            float av = pa[r][c+k], bv = pb[r][c+k], wk = w[k];
            s0 += wk*av; s1 += wk*bv; s2 += wk*av*av; s3 += wk*bv*bv; s4 += wk*av*bv;
        }
        hb[0][r][c]=s0; hb[1][r][c]=s1; hb[2][r][c]=s2; hb[3][r][c]=s3; hb[4][r][c]=s4;
    }
    __syncthreads();

    // vertical pass + SSIM
    int r = threadIdx.x >> 4, c = threadIdx.x & 15;
    float m1=0.f, m2=0.f, sxx=0.f, syy=0.f, sxy=0.f;
    #pragma unroll
    for (int k = 0; k < 11; k++) {
        float wk = w[k];
        m1 += wk*hb[0][r+k][c]; m2 += wk*hb[1][r+k][c];
        sxx += wk*hb[2][r+k][c]; syy += wk*hb[3][r+k][c]; sxy += wk*hb[4][r+k][c];
    }
    const float C1 = 1e-4f, C2 = 9e-4f;
    float mu1s=m1*m1, mu2s=m2*m2, mu12=m1*m2;
    float s1=sxx-mu1s, s2=syy-mu2s, s12=sxy-mu12;
    float val = ((2.f*mu12 + C1)*(2.f*s12 + C2)) / ((mu1s + mu2s + C1)*(s1 + s2 + C2));
    bool ok = (oy + r < OUTSZ) && (ox + c < OUTSZ);
    float contrib = ok ? val : 0.f;

    __shared__ float sb[256];
    float v = blockReduceSum(contrib, sb);
    if (!threadIdx.x) atomicAdd(&ws[3 + pair], v);
}

// ---------------- sliced Wasserstein: projection histograms ----------------
__global__ __launch_bounds__(256) void hist_kernel(const float* y, const float* mask, float* ws) {
    int b = blockIdx.y, pg = blockIdx.z;          // 8 projections per block
    float Sy = ws[SY_OFF+b] + EPSF, Sq = ws[SQ_OFF+b] + EPSF;
    float invSy = 1.f/Sy, invSq = 1.f/Sq;

    __shared__ float hP[8][NBIN];
    __shared__ float hQ[8][NBIN];
    for (int i = threadIdx.x; i < 8*NBIN; i += 256) { (&hP[0][0])[i] = 0.f; (&hQ[0][0])[i] = 0.f; }
    __syncthreads();

    float cs[8], sn[8];
    #pragma unroll
    for (int j = 0; j < 8; j++) {
        float t = 0.618034f*(float)(pg*8 + j + 1);
        t = (t - floorf(t))*6.2831853f;
        __sincosf(t, &sn[j], &cs[j]);
    }
    const float BSCALE = (float)NBIN/2.82843f;   // bins over [-sqrt2, sqrt2]
    for (int k = 0; k < 16; k++) {
        int i = blockIdx.x*4096 + k*256 + threadIdx.x;
        int r = i >> 9, c = i & 511;
        size_t idx = (size_t)b*HWSZ + i;
        float xv = (float)c*(2.f/511.f) - 1.f;
        float yv = (float)r*(2.f/511.f) - 1.f;
        float pval = (fmaxf(y[idx], 0.f) + EPSF)*invSy;
        float qval = (fminf(fmaxf(mask[idx], 0.f), 1.f) + EPSF)*invSq;
        #pragma unroll
        for (int j = 0; j < 8; j++) {
            float t = xv*cs[j] + yv*sn[j];
            int bin = (int)((t + 1.41422f)*BSCALE);
            bin = max(0, min(NBIN-1, bin));
            atomicAdd(&hP[j][bin], pval);
            atomicAdd(&hQ[j][bin], qval);
        }
    }
    __syncthreads();
    for (int i = threadIdx.x; i < 8*NBIN; i += 256) {
        int j = i >> 9, bin = i & 511;
        size_t g = (size_t)(b*NPROJ + pg*8 + j)*NBIN + bin;
        atomicAdd(&ws[HISTP_OFF + g], (&hP[0][0])[i]);
        atomicAdd(&ws[HISTQ_OFF + g], (&hQ[0][0])[i]);
    }
}

__global__ __launch_bounds__(512) void sw_reduce_kernel(float* ws) {
    int pid = blockIdx.x, b = blockIdx.y, t = threadIdx.x;
    size_t g = (size_t)(b*NPROJ + pid)*NBIN + t;
    __shared__ float sp[NBIN];
    __shared__ float sq[NBIN];
    sp[t] = ws[HISTP_OFF + g];
    sq[t] = ws[HISTQ_OFF + g];
    __syncthreads();
    for (int off = 1; off < NBIN; off <<= 1) {          // inclusive scan
        float a = (t >= off) ? sp[t-off] : 0.f;
        float c = (t >= off) ? sq[t-off] : 0.f;
        __syncthreads();
        sp[t] += a; sq[t] += c;
        __syncthreads();
    }
    const float DT = 2.82843f/(float)NBIN;
    float diff = fabsf(sp[t] - sq[t])*DT;
    __syncthreads();
    sp[t] = diff; __syncthreads();
    for (int s = NBIN >> 1; s > 0; s >>= 1) { if (t < s) sp[t] += sp[t+s]; __syncthreads(); }
    if (!t) atomicAdd(&ws[8], sp[0]);
}

// ---------------- Sinkhorn: sampling ----------------
__global__ __launch_bounds__(512) void sample_kernel(const float* y, const float* mask, float* ws) {
    int side = blockIdx.x, b = blockIdx.y, k = threadIdx.x;
    unsigned h = hash_u((unsigned)(b*131071 + k*8191 + side*524287 + 0x9E3779B9u));
    int idx = (int)(h & (HWSZ - 1));
    int r = idx >> 9, c = idx & 511;
    float xv = (float)c*(2.f/511.f) - 1.f;
    float yv = (float)r*(2.f/511.f) - 1.f;
    size_t gidx = (size_t)b*HWSZ + idx;
    float pv;
    if (side == 0) pv = (fmaxf(y[gidx], 0.f) + EPSF)/(ws[SY_OFF+b] + EPSF);
    else           pv = (fminf(fmaxf(mask[gidx], 0.f), 1.f) + EPSF)/(ws[SQ_OFF+b] + EPSF);
    __shared__ float sb[512];
    float tot = blockReduceSum(pv, sb);
    float wv = pv/(tot + EPSF);
    size_t co = (size_t)((side ? XT_OFF : XS_OFF)) + ((size_t)b*KS + k)*2;
    ws[co]   = xv;
    ws[co+1] = yv;
    ws[(side ? WB_OFF : WA_OFF) + (size_t)b*KS + k] = wv;
}

// ---------------- Sinkhorn cost matrix via WMMA f16 (K padded 2->32) ----------------
__global__ __launch_bounds__(32) void cost_kernel(float* ws) {
    int b = blockIdx.z, ti = blockIdx.y, tj0 = blockIdx.x*4;
    int lane = threadIdx.x;
    const float* XS = ws + XS_OFF + (size_t)b*KS*2;
    const float* XT = ws + XT_OFF + (size_t)b*KS*2;
    float* Cb = ws + C_OFF + (size_t)b*KS*KS;

    __shared__ float sx[16], sy[16], tx[64], ty[64];
    if (lane < 16) { int rr = ti*16 + lane; sx[lane] = XS[rr*2]; sy[lane] = XS[rr*2 + 1]; }
    {
        int c0 = tj0*16 + lane;       tx[lane]    = XT[c0*2]; ty[lane]    = XT[c0*2 + 1];
        int c1 = c0 + 32;             tx[lane+32] = XT[c1*2]; ty[lane+32] = XT[c1*2 + 1];
    }
    __syncthreads();

    // A (16x32 f16): lane<16 holds row M=lane, K=0..7 in low halves -> only K=0,1 nonzero
    v16h a = {};
    if (lane < 16) { a[0] = (_Float16)sx[lane]; a[1] = (_Float16)sy[lane]; }

    #pragma unroll
    for (int u = 0; u < 4; u++) {
        v16h bb = {};
        if (lane < 16) { bb[0] = (_Float16)tx[u*16 + lane]; bb[1] = (_Float16)ty[u*16 + lane]; }
        v8f acc = {};
        acc = __builtin_amdgcn_wmma_f32_16x16x32_f16(false, a, false, bb,
                                                     (short)0, acc, false, false);
        int n = lane & 15;
        int mbase = (lane < 16) ? 0 : 8;
        float xn2 = tx[u*16+n]*tx[u*16+n] + ty[u*16+n]*ty[u*16+n];
        #pragma unroll
        for (int r = 0; r < 8; r++) {
            int m = mbase + r;
            float xm2 = sx[m]*sx[m] + sy[m]*sy[m];
            float Cv = fmaxf(xm2 + xn2 - 2.f*acc[r], 0.f);
            Cb[(size_t)(ti*16 + m)*KS + (tj0+u)*16 + n] = Cv;
        }
    }
}

// ---------------- Sinkhorn log-domain iterations (C resident in L2) ----------------
__global__ __launch_bounds__(512) void sinkhorn_kernel(float* ws) {
    int b = blockIdx.x, i = threadIdx.x;
    const float* Cb = ws + C_OFF + (size_t)b*KS*KS;
    __shared__ float lu[KS], lv[KS], la[KS], lb[KS];
    la[i] = __logf(ws[WA_OFF + (size_t)b*KS + i] + 1e-12f);
    lb[i] = __logf(ws[WB_OFF + (size_t)b*KS + i] + 1e-12f);
    lu[i] = 0.f; lv[i] = 0.f;
    __syncthreads();
    const float* row = Cb + (size_t)i*KS;
    for (int it = 0; it < 50; ++it) {
        float m = -3.0e38f, s = 0.f;
        __builtin_prefetch(row, 0, 1);
        for (int j = 0; j < KS; j++) {
            float z = -20.f*row[j] + lv[j];
            if (z > m) { s = s*__expf(m - z) + 1.f; m = z; } else s += __expf(z - m);
        }
        lu[i] = la[i] - (m + __logf(s));
        __syncthreads();
        m = -3.0e38f; s = 0.f;
        for (int j = 0; j < KS; j++) {
            float z = -20.f*Cb[(size_t)j*KS + i] + lu[j];
            if (z > m) { s = s*__expf(m - z) + 1.f; m = z; } else s += __expf(z - m);
        }
        lv[i] = lb[i] - (m + __logf(s));
        __syncthreads();
    }
    float acc = 0.f;
    for (int j = 0; j < KS; j++) {
        float cv = row[j];
        float lp = fminf(lu[i] - 20.f*cv + lv[j], 30.f);
        acc += __expf(lp)*cv;
    }
    __shared__ float sb[512];
    float v = blockReduceSum(acc, sb);
    if (!i) atomicAdd(&ws[9], v);
}

// ---------------- final combine ----------------
__global__ void final_kernel(const float* ws, float* out) {
    if (threadIdx.x == 0 && blockIdx.x == 0) {
        const float invN = 1.f/(float)NTOT;
        const float cnt_ssim = 16.f*502.f*502.f;
        float ssim_m = 1.f - ws[3]/cnt_ssim;
        float ssim_p = 1.f - ws[4]/cnt_ssim;
        float l1m = ws[0]*invN, l1p = ws[1]*invN, petsal = ws[2]*invN;
        float g   = ws[5]*invN + ws[6]*(4.f*invN);
        float aux = ws[7]*invN;
        float sot = ws[8]/((float)NPROJ*(float)BB);
        float sink = ws[9]/(float)BB;
        out[0] = 2.5f*ssim_m + 2.0f*ssim_p + 1.0f*l1m + 1.0f*l1p
               + 2.0f*g + 1.0f*aux + 3.0f*petsal + 2.0f*sot + 2.0f*sink;
    }
}

// ---------------- launch ----------------
extern "C" void kernel_launch(void* const* d_in, const int* in_sizes, int n_in,
                              void* d_out, int out_size, void* d_ws, size_t ws_size,
                              hipStream_t stream) {
    const float* y    = (const float*)d_in[0];
    const float* mri  = (const float*)d_in[1];
    const float* pet  = (const float*)d_in[2];
    const float* mask = (const float*)d_in[3];
    const float* ehat = (const float*)d_in[4];
    float* out = (float*)d_out;
    float* ws  = (float*)d_ws;
    (void)in_sizes; (void)n_in; (void)out_size; (void)ws_size;

    init_kernel<<<dim3(1024), dim3(256), 0, stream>>>(ws);
    passA_full_kernel<<<dim3(64, BB), dim3(256), 0, stream>>>(y, mri, pet, mask, ws);
    passA_half_kernel<<<dim3(16, BB), dim3(256), 0, stream>>>(y, mri, ws);
    ssim_kernel<<<dim3(32, 32, BB*2), dim3(256), 0, stream>>>(y, mri, pet, ws);
    passB_full_kernel<<<dim3(64, BB), dim3(256), 0, stream>>>(y, mri, pet, ehat, ws);
    passB_half_kernel<<<dim3(16, BB), dim3(256), 0, stream>>>(y, mri, ws);
    hist_kernel<<<dim3(64, BB, 8), dim3(256), 0, stream>>>(y, mask, ws);
    sw_reduce_kernel<<<dim3(NPROJ, BB), dim3(512), 0, stream>>>(ws);
    sample_kernel<<<dim3(2, BB), dim3(512), 0, stream>>>(y, mask, ws);
    cost_kernel<<<dim3(8, 32, BB), dim3(32), 0, stream>>>(ws);
    sinkhorn_kernel<<<dim3(BB), dim3(512), 0, stream>>>(ws);
    final_kernel<<<1, 1, 0, stream>>>(ws, out);
}
